// Tx8MixtralSparseMoeBlock_31413390803057
// MI455X (gfx1250) — matrix operand order repaired and checked by reading, not compiled, not run on hardware
//
#include <hip/hip_runtime.h>
#include <hip/hip_bf16.h>
#include <math.h>

// Problem constants (B=1)
constexpr int T = 2048;   // tokens = B*S
constexpr int H = 1024;   // hidden
constexpr int F = 3584;   // ffn dim
constexpr int E = 8;      // experts

typedef __attribute__((ext_vector_type(16))) __bf16 v16bf;
typedef __attribute__((ext_vector_type(8)))  __bf16 v8bf;
typedef __attribute__((ext_vector_type(8)))  float  v8f;

__device__ __forceinline__ v8f zero8() {
  v8f z;
#pragma unroll
  for (int i = 0; i < 8; ++i) z[i] = 0.0f;
  return z;
}

__device__ __forceinline__ v8f wmma_bf16(v16bf a, v16bf b, v8f c) {
  // D = A(16x32 bf16) * B(32x16 bf16) + C(16x16 f32)
  return __builtin_amdgcn_wmma_f32_16x16x32_bf16(
      /*neg_a=*/false, a, /*neg_b=*/false, b,
      /*c_mod=*/(short)0, c, /*reuse_a=*/false, /*reuse_b=*/false);
}

// Load A fragment (16 rows x 32 K, bf16) from an LDS tile with row stride
// `stride` (elements). Layout: lanes 0-15 hold rows M=0..15 with
// K = [0..15]; lanes 16-31 hold K = [16..31] (matches contiguous B frags).
// Loaded as two 16B ds loads.
__device__ __forceinline__ v16bf load_a_frag(const __bf16* base, int stride,
                                             int k0, int lane) {
  const int m  = lane & 15;
  const int kb = (lane >> 4) << 4;     // 0 or 16
  const __bf16* p = base + m * stride + k0 + kb;
  v8bf lo = *(const v8bf*)(p);
  v8bf hi = *(const v8bf*)(p + 8);
  v16bf a;
#pragma unroll
  for (int i = 0; i < 8; ++i) { a[i] = lo[i]; a[8 + i] = hi[i]; }
  return a;
}

// ---------------------------------------------------------------------------
// Kernel 1: router. One wave32 per token. Computes logits (written to d_out's
// logits section as bf16), fp32 softmax over bf16-rounded logits, top-2,
// renormalized weights, and appends (token, coef) to per-expert lists.
// ---------------------------------------------------------------------------
__global__ void __launch_bounds__(256)
router_kernel(const __bf16* __restrict__ x, const __bf16* __restrict__ gw,
              __bf16* __restrict__ logits_out, int* __restrict__ counts,
              int* __restrict__ etok, float* __restrict__ ecoef) {
  const int wv = threadIdx.x >> 5;
  const int lane = threadIdx.x & 31;
  const int t = blockIdx.x * 8 + wv;
  if (t >= T) return;

  float p[E];
#pragma unroll
  for (int e = 0; e < E; ++e) p[e] = 0.0f;

  const __bf16* xr = x + (size_t)t * H;
  for (int h = lane; h < H; h += 32) {
    const float xv = (float)xr[h];
#pragma unroll
    for (int e = 0; e < E; ++e) p[e] += xv * (float)gw[(size_t)e * H + h];
  }
#pragma unroll
  for (int e = 0; e < E; ++e) {
#pragma unroll
    for (int off = 16; off >= 1; off >>= 1) p[e] += __shfl_xor(p[e], off, 32);
  }

  // bf16 logits (reference does bf16 cast before softmax)
  float le[E];
#pragma unroll
  for (int e = 0; e < E; ++e) {
    __bf16 lb = (__bf16)p[e];
    le[e] = (float)lb;
    if (lane == 0) logits_out[(size_t)t * E + e] = lb;
  }

  if (lane != 0) return;

  float mx = le[0];
#pragma unroll
  for (int e = 1; e < E; ++e) mx = fmaxf(mx, le[e]);
  float pr[E]; float s = 0.0f;
#pragma unroll
  for (int e = 0; e < E; ++e) { pr[e] = __expf(le[e] - mx); s += pr[e]; }
  const float inv = 1.0f / s;
#pragma unroll
  for (int e = 0; e < E; ++e) pr[e] = (float)((__bf16)(pr[e] * inv));  // bf16 probs

  int i1 = 0; float m1 = pr[0];
#pragma unroll
  for (int e = 1; e < E; ++e) if (pr[e] > m1) { m1 = pr[e]; i1 = e; }
  int i2 = (i1 == 0) ? 1 : 0; float m2 = pr[i2];
#pragma unroll
  for (int e = 0; e < E; ++e)
    if (e != i1 && pr[e] > m2) { m2 = pr[e]; i2 = e; }

  const float denom = 1.0f / (m1 + m2);
  const float w1 = (float)((__bf16)(m1 * denom));
  const float w2 = (float)((__bf16)(m2 * denom));

  int s1 = atomicAdd(&counts[i1], 1);
  etok[(size_t)i1 * T + s1] = t; ecoef[(size_t)i1 * T + s1] = w1;
  int s2 = atomicAdd(&counts[i2], 1);
  etok[(size_t)i2 * T + s2] = t; ecoef[(size_t)i2 * T + s2] = w2;
}

// ---------------------------------------------------------------------------
// Kernel 2: fused expert MLP. One block (8 waves) per (expert, 32-token tile).
// out_partial[32,H] += coef * ( silu(X Wg^T) * (X Wu^T) ) Wd^T
// M=32: each wave carries two 16-row A-fragments per B fragment, halving the
// expert-weight traffic vs M=16. F is processed in 128-wide chunks; the 32x1024
// down-proj accumulators persist in VGPRs (16 v8f frags per wave).
// ---------------------------------------------------------------------------
constexpr int MT      = 32;       // tokens per block tile
constexpr int TILES   = T / MT;   // 64 token tiles per expert (upper bound)
constexpr int FC      = 128;      // F chunk width
constexpr int NCHUNK  = F / FC;   // 28
constexpr int XS_STR  = H + 8;    // LDS row stride (keeps 16B alignment)
constexpr int IS_STR  = FC + 8;

constexpr size_t MOE_LDS_BYTES =
    (size_t)MT * XS_STR * 2 +     // Xs
    (size_t)MT * IS_STR * 2 +     // interS
    MT * sizeof(int) +            // tokS
    MT * sizeof(float);           // coefS

__global__ void __launch_bounds__(256)
moe_kernel(const __bf16* __restrict__ x,
           const __bf16* __restrict__ w_gate,
           const __bf16* __restrict__ w_up,
           const __bf16* __restrict__ w_down,
           const int* __restrict__ counts,
           const int* __restrict__ etok,
           const float* __restrict__ ecoef,
           float* __restrict__ accum) {
  const int e    = blockIdx.x / TILES;
  const int tile = blockIdx.x % TILES;
  const int cnt  = counts[e];
  if (tile * MT >= cnt) return;

  extern __shared__ __align__(16) char smem[];
  __bf16* Xs     = (__bf16*)smem;                 // [MT][XS_STR]
  __bf16* interS = Xs + MT * XS_STR;              // [MT][IS_STR]
  int*    tokS   = (int*)(interS + MT * IS_STR);  // [MT]
  float*  coefS  = (float*)(tokS + MT);           // [MT]

  const int tid  = threadIdx.x;
  const int wv   = tid >> 5;
  const int lane = tid & 31;

  if (tid < MT) {
    const int slot = tile * MT + tid;
    if (slot < cnt) {
      tokS[tid]  = etok[(size_t)e * T + slot];
      coefS[tid] = ecoef[(size_t)e * T + slot];
    } else {
      tokS[tid]  = 0;       // pad: valid row, zero coef
      coefS[tid] = 0.0f;
    }
  }
  __syncthreads();

  // Gather MT token rows of X into LDS (16B vector copies).
  for (int c = tid; c < MT * (H / 8); c += 256) {
    const int row  = c >> 7;            // H/8 = 128 chunks per row
    const int col8 = (c & 127) << 3;
    *(uint4*)&Xs[row * XS_STR + col8] =
        *(const uint4*)&x[(size_t)tokS[row] * H + col8];
  }

  const __bf16* wgE = w_gate + (size_t)e * F * H;
  const __bf16* wuE = w_up   + (size_t)e * F * H;
  const __bf16* wdE = w_down + (size_t)e * H * F;

  const int n    = lane & 15;
  const int kb16 = (lane >> 4) << 4;    // 0 or 16 (B-frag K group)
  const int hi   = lane >> 4;

  // Persistent down-proj accumulators: accd[2*j + mf] covers
  // h-tile (wv*8+j), token rows [mf*16, mf*16+16).
  v8f accd[16];
#pragma unroll
  for (int j = 0; j < 16; ++j) accd[j] = zero8();

  for (int c = 0; c < NCHUNK; ++c) {
    const int fbase = c * FC;
    __syncthreads();  // interS free for reuse; Xs ready on first iter

    // --- gate & up projections for this wave's 16-wide f-tile (both m-frags)
    v8f ag0 = zero8(), au0 = zero8(), ag1 = zero8(), au1 = zero8();
    const int f = fbase + wv * 16 + n;
    const __bf16* wgRow = wgE + (size_t)f * H;
    const __bf16* wuRow = wuE + (size_t)f * H;
    if (c + 1 < NCHUNK) {  // warm GL2 for next chunk's weight rows
      __builtin_prefetch(wgRow + (size_t)FC * H, 0, 1);
      __builtin_prefetch(wuRow + (size_t)FC * H, 0, 1);
    }
#pragma unroll 2
    for (int k0 = 0; k0 < H; k0 += 32) {
      v16bf a0 = load_a_frag(Xs, XS_STR, k0, lane);
      v16bf a1 = load_a_frag(Xs + 16 * XS_STR, XS_STR, k0, lane);
      v16bf bg = *(const v16bf*)(wgRow + k0 + kb16);
      v16bf bu = *(const v16bf*)(wuRow + k0 + kb16);
      ag0 = wmma_bf16(a0, bg, ag0);
      ag1 = wmma_bf16(a1, bg, ag1);
      au0 = wmma_bf16(a0, bu, au0);
      au1 = wmma_bf16(a1, bu, au1);
    }
    // silu(g) * u -> bf16 intermediate chunk in LDS (v_rcp, not precise div)
#pragma unroll
    for (int r = 0; r < 8; ++r) {
      const float g0 = ag0[r], u0 = au0[r];
      const float g1 = ag1[r], u1 = au1[r];
      const float s0 = g0 * __builtin_amdgcn_rcpf(1.0f + __expf(-g0));
      const float s1 = g1 * __builtin_amdgcn_rcpf(1.0f + __expf(-g1));
      const int rr = r + 8 * hi;
      interS[rr * IS_STR + wv * 16 + n]        = (__bf16)(s0 * u0);
      interS[(16 + rr) * IS_STR + wv * 16 + n] = (__bf16)(s1 * u1);
    }
    __syncthreads();

    // --- fold chunk into persistent down-proj accumulators ---
#pragma unroll
    for (int j = 0; j < 8; ++j) {
      const int h0 = (wv * 8 + j) << 4;
      const __bf16* wdRow = wdE + (size_t)(h0 + n) * F + fbase + kb16;
#pragma unroll
      for (int kk = 0; kk < FC; kk += 32) {
        v16bf a0 = load_a_frag(interS, IS_STR, kk, lane);
        v16bf a1 = load_a_frag(interS + 16 * IS_STR, IS_STR, kk, lane);
        v16bf bd = *(const v16bf*)(wdRow + kk);
        accd[2 * j + 0] = wmma_bf16(a0, bd, accd[2 * j + 0]);
        accd[2 * j + 1] = wmma_bf16(a1, bd, accd[2 * j + 1]);
      }
    }
  }

  // Scale by routing coef and combine (fp32 atomics; 2 contributions/elem).
#pragma unroll
  for (int j = 0; j < 8; ++j) {
    const int h0 = (wv * 8 + j) << 4;
#pragma unroll
    for (int mf = 0; mf < 2; ++mf) {
#pragma unroll
      for (int r = 0; r < 8; ++r) {
        const int m = mf * 16 + r + 8 * hi;
        const float v = accd[2 * j + mf][r] * coefS[m];
        atomicAdd(&accum[(size_t)tokS[m] * H + h0 + n], v);
      }
    }
  }
}

// ---------------------------------------------------------------------------
// Kernel 3: fp32 accumulator -> bf16 output
// ---------------------------------------------------------------------------
__global__ void __launch_bounds__(256)
convert_kernel(const float* __restrict__ acc, __bf16* __restrict__ out, int n) {
  int i = (blockIdx.x * blockDim.x + threadIdx.x) * 4;
  if (i + 3 < n) {
    float4 v = *(const float4*)&acc[i];
    out[i + 0] = (__bf16)v.x; out[i + 1] = (__bf16)v.y;
    out[i + 2] = (__bf16)v.z; out[i + 3] = (__bf16)v.w;
  } else {
    for (; i < n; ++i) out[i] = (__bf16)acc[i];
  }
}

extern "C" void kernel_launch(void* const* d_in, const int* in_sizes, int n_in,
                              void* d_out, int out_size, void* d_ws,
                              size_t ws_size, hipStream_t stream) {
  (void)in_sizes; (void)n_in; (void)out_size; (void)ws_size;
  const __bf16* x  = (const __bf16*)d_in[0];   // [T,H]
  const __bf16* gw = (const __bf16*)d_in[1];   // [E,H]
  const __bf16* wg = (const __bf16*)d_in[2];   // [E,F,H]
  const __bf16* wu = (const __bf16*)d_in[3];   // [E,F,H]
  const __bf16* wd = (const __bf16*)d_in[4];   // [E,H,F]

  __bf16* out    = (__bf16*)d_out;             // [T,H] bf16
  __bf16* logits = out + (size_t)T * H;        // [T,E] bf16 (tuple output #2)

  // Workspace: fp32 accum [T,H] | counts[32] | etok[E,T] | ecoef[E,T]
  float* accum = (float*)d_ws;
  int*   counts = (int*)((char*)d_ws + (size_t)T * H * sizeof(float));
  int*   etok   = counts + 32;
  float* ecoef  = (float*)(etok + (size_t)E * T);

  hipMemsetAsync(accum, 0, (size_t)T * H * sizeof(float), stream);
  hipMemsetAsync(counts, 0, 32 * sizeof(int), stream);

  router_kernel<<<T / 8, 256, 0, stream>>>(x, gw, logits, counts, etok, ecoef);
  moe_kernel<<<E * TILES, 256, MOE_LDS_BYTES, stream>>>(x, wg, wu, wd, counts,
                                                        etok, ecoef, accum);
  const int n = T * H;
  convert_kernel<<<(n / 4 + 255) / 256, 256, 0, stream>>>(accum, out, n);
}
